// SimpleRNN_13864154432191
// MI455X (gfx1250) — compile-verified, matrix operand order checked by reading, not run on hardware
//
#include <hip/hip_runtime.h>
#include <hip/hip_bf16.h>
#include <math.h>

typedef __attribute__((ext_vector_type(16))) _Float16 v16h;
typedef __attribute__((ext_vector_type(8)))  _Float16 v8h;
typedef __attribute__((ext_vector_type(8)))  float    v8f;

#define B_ 512
#define T_ 512
#define F_ 13
#define H_ 128

// ---------------------------------------------------------------------------
// Kernel 1: fused input projection + recurrent scan.
// grid = B/16 workgroups, block = 256 threads (8 waves).
// Wave w owns hidden columns [16w, 16w+16); its 4 W_hh B-fragments live in
// VGPRs for the whole 512-step scan. h cycles through LDS each step.
// Emits hs[B*T, H] in f16 for the head GEMM.
// ---------------------------------------------------------------------------
__global__ __launch_bounds__(256)
void rnn_scan_kernel(const float* __restrict__ x,      // [B,T,F]
                     const float* __restrict__ W_ih,   // [H,F]
                     const float* __restrict__ W_hh,   // [H,H]
                     const float* __restrict__ b_ih,   // [H]
                     const float* __restrict__ b_hh,   // [H]
                     _Float16* __restrict__ hs)        // [B*T, H] f16
{
    __shared__ float    wih_lds[H_ * F_];
    __shared__ float    bias_lds[H_];
    __shared__ float    x_lds[16 * F_];
    __shared__ float    xp_lds[16][H_];
    __shared__ _Float16 h_lds[16][H_];

    const int tid  = threadIdx.x;
    const int lane = tid & 31;
    const int w    = tid >> 5;        // wave 0..7
    const int hi   = lane >> 4;       // half-wave select
    const int nl   = lane & 15;       // local column / row-in-tile
    const int b0   = blockIdx.x * 16; // batch tile base

    // ---- one-time setup -------------------------------------------------
    for (int i = tid; i < H_ * F_; i += 256) wih_lds[i]  = W_ih[i];
    for (int i = tid; i < H_;      i += 256) bias_lds[i] = b_ih[i] + b_hh[i];
    for (int i = tid; i < 16 * H_; i += 256) (&h_lds[0][0])[i] = (_Float16)0.f;

    // B-fragments for h @ W_hh^T:  B[k][n] = W_hh[n][k], n = 16w + nl.
    // 16-bit B layout mirrors A: halves 0..7 -> K = k0+8*hi+j,
    //                            halves 8..15 -> K = k0+16+8*hi+(j-8).
    v16h bfrag[4];
    {
        const int n = 16 * w + nl;
        #pragma unroll
        for (int kb = 0; kb < 4; ++kb) {
            const int k0 = 32 * kb;
            v16h f;
            #pragma unroll
            for (int j = 0; j < 16; ++j) {
                const int k = (j < 8) ? (k0 + 8 * hi + j)
                                      : (k0 + 16 + 8 * hi + (j - 8));
                f[j] = (_Float16)W_hh[n * H_ + k];
            }
            bfrag[kb] = f;
        }
    }
    __syncthreads();

    // ---- time scan ------------------------------------------------------
    for (int t = 0; t < T_; ++t) {
        // stage the 16 x-rows (13 floats each) for this timestep
        if (tid < 16 * F_) {
            const int m = tid / F_, f = tid - m * F_;
            x_lds[tid] = x[(((size_t)(b0 + m)) * T_ + t) * F_ + f];
        }
        // prefetch next timestep's rows into L2 (global_prefetch_b8)
        if (t + 1 < T_ && tid < 16)
            __builtin_prefetch(&x[(((size_t)(b0 + tid)) * T_ + (t + 1)) * F_], 0, 1);
        __syncthreads();   // x ready; h(t-1) writes visible

        // xp tile [16][128]: 8 outputs per thread, K=13 dot from LDS
        {
            const int obase = tid * 8;
            const int m = obase >> 7;
            const int n = obase & 127;
            #pragma unroll
            for (int i = 0; i < 8; ++i) {
                float acc = bias_lds[n + i];
                #pragma unroll
                for (int f = 0; f < F_; ++f)
                    acc += x_lds[m * F_ + f] * wih_lds[(n + i) * F_ + f];
                xp_lds[m][n + i] = acc;
            }
        }

        // A-fragments of h(t-1) from LDS (two ds_load_b128 per fragment)
        v16h afrag[4];
        {
            const int m = nl;
            #pragma unroll
            for (int kb = 0; kb < 4; ++kb) {
                const int k0 = 32 * kb + 8 * hi;
                const v8h a0 = *reinterpret_cast<const v8h*>(&h_lds[m][k0]);
                const v8h a1 = *reinterpret_cast<const v8h*>(&h_lds[m][k0 + 16]);
                v16h f;
                #pragma unroll
                for (int j = 0; j < 8; ++j) { f[j] = a0[j]; f[j + 8] = a1[j]; }
                afrag[kb] = f;
            }
        }
        __syncthreads();   // xp ready; old-h reads complete before overwrite

        // 16x16 tile of h(t) = A(16x128) x B(128x16), f32 accumulate
        v8f c = {};
        #pragma unroll
        for (int kb = 0; kb < 4; ++kb)
            c = __builtin_amdgcn_wmma_f32_16x16x32_f16(
                    false, afrag[kb], false, bfrag[kb],
                    (short)0, c, false, false);

        // epilogue: +xp, tanh, write back to LDS and global hs
        const int n = 16 * w + nl;
        #pragma unroll
        for (int r = 0; r < 8; ++r) {
            const int m = r + 8 * hi;
            const float hv = tanhf(c[r] + xp_lds[m][n]);
            const _Float16 hh = (_Float16)hv;
            h_lds[m][n] = hh;
            hs[(((size_t)(b0 + m)) * T_ + t) * H_ + n] = hh;
        }
        // no trailing barrier needed: next iteration's first barrier orders
        // these h/xp accesses against the next step's writes.
    }
}

// ---------------------------------------------------------------------------
// Kernel 2: head  y = sigmoid(relu(hs @ fc1_w^T + fc1_b) @ fc2_w^T + fc2_b)
// grid = (B*T)/16 workgroups, block = 128 (4 waves).
// Wave w computes the 16x16 tile of fc1 columns [16w,16w+16) with WMMA.
// ---------------------------------------------------------------------------
__global__ __launch_bounds__(128)
void head_kernel(const _Float16* __restrict__ hs,    // [B*T, H]
                 const float* __restrict__ fc1_w,    // [64, H]
                 const float* __restrict__ fc1_b,    // [64]
                 const float* __restrict__ fc2_w,    // [64]
                 const float* __restrict__ fc2_b,    // [1]
                 float* __restrict__ out)            // [B*T]
{
    __shared__ float y1_lds[16][64];
    __shared__ float fc2w_lds[64];

    const int tid  = threadIdx.x;
    const int lane = tid & 31;
    const int w    = tid >> 5;   // 0..3
    const int hi   = lane >> 4;
    const int nl   = lane & 15;
    const size_t r0 = (size_t)blockIdx.x * 16;

    if (tid < 64) fc2w_lds[tid] = fc2_w[tid];

    // B-fragments: B[k][n] = fc1_w[n][k], n = 16w + nl
    const int n = 16 * w + nl;
    const float bias = fc1_b[n];
    v16h bfrag[4];
    #pragma unroll
    for (int kb = 0; kb < 4; ++kb) {
        const int k0 = 32 * kb;
        v16h f;
        #pragma unroll
        for (int j = 0; j < 16; ++j) {
            const int k = (j < 8) ? (k0 + 8 * hi + j)
                                  : (k0 + 16 + 8 * hi + (j - 8));
            f[j] = (_Float16)fc1_w[n * H_ + k];
        }
        bfrag[kb] = f;
    }

    // A-fragments straight from global hs (global_load_b128 pairs)
    v16h afrag[4];
    {
        const _Float16* row = hs + (r0 + nl) * H_;
        #pragma unroll
        for (int kb = 0; kb < 4; ++kb) {
            const int k0 = 32 * kb + 8 * hi;
            const v8h a0 = *reinterpret_cast<const v8h*>(row + k0);
            const v8h a1 = *reinterpret_cast<const v8h*>(row + k0 + 16);
            v16h f;
            #pragma unroll
            for (int j = 0; j < 8; ++j) { f[j] = a0[j]; f[j + 8] = a1[j]; }
            afrag[kb] = f;
        }
    }

    v8f c = {};
    #pragma unroll
    for (int kb = 0; kb < 4; ++kb)
        c = __builtin_amdgcn_wmma_f32_16x16x32_f16(
                false, afrag[kb], false, bfrag[kb],
                (short)0, c, false, false);

    // bias + relu into LDS
    #pragma unroll
    for (int r = 0; r < 8; ++r) {
        const int m = r + 8 * hi;
        const float v = c[r] + bias;
        y1_lds[m][n] = v > 0.f ? v : 0.f;
    }
    __syncthreads();

    // fc2 (64-dot) + sigmoid, one lane per output row
    if (tid < 16) {
        float z = fc2_b[0];
        #pragma unroll
        for (int o = 0; o < 64; ++o)
            z += y1_lds[tid][o] * fc2w_lds[o];
        out[r0 + tid] = 1.f / (1.f + expf(-z));
    }
}

// ---------------------------------------------------------------------------
extern "C" void kernel_launch(void* const* d_in, const int* in_sizes, int n_in,
                              void* d_out, int out_size, void* d_ws, size_t ws_size,
                              hipStream_t stream) {
    (void)in_sizes; (void)n_in; (void)out_size; (void)ws_size;
    const float* x     = (const float*)d_in[0];
    const float* W_ih  = (const float*)d_in[1];
    const float* W_hh  = (const float*)d_in[2];
    const float* b_ih  = (const float*)d_in[3];
    const float* b_hh  = (const float*)d_in[4];
    const float* fc1_w = (const float*)d_in[5];
    const float* fc1_b = (const float*)d_in[6];
    const float* fc2_w = (const float*)d_in[7];
    const float* fc2_b = (const float*)d_in[8];
    float* out = (float*)d_out;

    _Float16* hs = (_Float16*)d_ws;   // needs B*T*H*2 = 64 MB of workspace

    rnn_scan_kernel<<<B_ / 16, 256, 0, stream>>>(x, W_ih, W_hh, b_ih, b_hh, hs);
    head_kernel<<<(B_ * T_) / 16, 128, 0, stream>>>(hs, fc1_w, fc1_b, fc2_w,
                                                    fc2_b, out);
}
